// RESModule_28389733826887
// MI455X (gfx1250) — matrix operand order repaired and checked by reading, not compiled, not run on hardware
//
#include <hip/hip_runtime.h>
#include <hip/hip_bf16.h>

typedef __attribute__((ext_vector_type(16))) _Float16 v16h;
typedef __attribute__((ext_vector_type(8)))  float    v8f;

#define B_GR 8
#define NPER 4096
#define KNN  16
#define DIN  32
#define DH   64
#define DOUT 128
#define NTOT (B_GR * NPER)          // 32768
#define NEDGE (NTOT * KNN)          // 524288
#define KSPLIT 4
#define QPB 64                      // queries per KNN block

// ---------------------------------------------------------------------------
// Kernel 1: brute-force per-graph KNN, 4-way split scan + LDS merge.
// Block = 64 queries x 4 splits; graph positions (48 KB) staged in LDS.
// ---------------------------------------------------------------------------
__global__ __launch_bounds__(256) void knn_kernel(const float* __restrict__ pos,
                                                  int* __restrict__ col) {
    __shared__ float sx[NPER], sy[NPER], sz[NPER];
    __shared__ float md[QPB * KSPLIT * KNN];
    __shared__ int   mi[QPB * KSPLIT * KNN];
    const int g  = blockIdx.x >> 6;          // graph (64 blocks per graph)
    const int qc = blockIdx.x & 63;
    const float* pg = pos + (size_t)g * NPER * 3;
    for (int i = threadIdx.x; i < NPER; i += 256) {
        sx[i] = pg[i * 3 + 0]; sy[i] = pg[i * 3 + 1]; sz[i] = pg[i * 3 + 2];
    }
    __syncthreads();

    const int ql = threadIdx.x >> 2, sp = threadIdx.x & 3;
    const int q  = qc * QPB + ql;
    const float qx = sx[q], qy = sy[q], qz = sz[q];

    float bd[KNN]; int bi[KNN];
#pragma unroll
    for (int j = 0; j < KNN; ++j) { bd[j] = 3.4e38f; bi[j] = 0; }

    const int c0 = sp * (NPER / KSPLIT);
    for (int cc = 0; cc < NPER / KSPLIT; ++cc) {
        const int c = c0 + cc;
        const float dx = qx - sx[c], dy = qy - sy[c], dz = qz - sz[c];
        const float d  = dx * dx + dy * dy + dz * dz;
        if (d < bd[KNN - 1]) {
            bd[KNN - 1] = d; bi[KNN - 1] = c;
#pragma unroll
            for (int j = KNN - 1; j > 0; --j)
                if (bd[j] < bd[j - 1]) {
                    float td = bd[j]; bd[j] = bd[j - 1]; bd[j - 1] = td;
                    int   ti = bi[j]; bi[j] = bi[j - 1]; bi[j - 1] = ti;
                }
        }
    }
    const int mb = (ql * KSPLIT + sp) * KNN;
#pragma unroll
    for (int j = 0; j < KNN; ++j) { md[mb + j] = bd[j]; mi[mb + j] = bi[j]; }
    __syncthreads();

    if (sp == 0) {
        for (int s2 = 1; s2 < KSPLIT; ++s2) {
            const int ob = (ql * KSPLIT + s2) * KNN;
#pragma unroll
            for (int j = 0; j < KNN; ++j) {
                const float d = md[ob + j]; const int ci = mi[ob + j];
                if (d < bd[KNN - 1]) {
                    bd[KNN - 1] = d; bi[KNN - 1] = ci;
#pragma unroll
                    for (int j2 = KNN - 1; j2 > 0; --j2)
                        if (bd[j2] < bd[j2 - 1]) {
                            float td = bd[j2]; bd[j2] = bd[j2 - 1]; bd[j2 - 1] = td;
                            int   ti = bi[j2]; bi[j2] = bi[j2 - 1]; bi[j2 - 1] = ti;
                        }
                }
            }
        }
        const int base = (g * NPER + q) * KNN;
#pragma unroll
        for (int j = 0; j < KNN; ++j) col[base + j] = g * NPER + bi[j];
    }
}

// ---------------------------------------------------------------------------
// Kernel 2: generic zero (16B granules)
// ---------------------------------------------------------------------------
__global__ void zero_kernel(uint4* __restrict__ p, int n) {
    const int i = blockIdx.x * 256 + threadIdx.x;
    if (i < n) p[i] = make_uint4(0u, 0u, 0u, 0u);
}

// ---------------------------------------------------------------------------
// CSR inversion of the edge->receiver map (replaces 33.5M f32 atomics)
// ---------------------------------------------------------------------------
__global__ void count_kernel(const int* __restrict__ col, int* __restrict__ cnt) {
    const int e = blockIdx.x * 256 + threadIdx.x;
    atomicAdd(&cnt[col[e]], 1);
}

__global__ __launch_bounds__(1024) void scan_kernel(const int* __restrict__ cnt,
                                                    int* __restrict__ ofs,
                                                    int* __restrict__ cur) {
    __shared__ int part[1024];
    const int t = threadIdx.x;
    int loc[32];
    int s = 0;
#pragma unroll
    for (int i = 0; i < 32; ++i) { loc[i] = cnt[t * 32 + i]; s += loc[i]; }
    part[t] = s;
    __syncthreads();
    for (int d = 1; d < 1024; d <<= 1) {           // Hillis-Steele inclusive
        const int v = part[t];
        const int u = (t >= d) ? part[t - d] : 0;
        __syncthreads();
        part[t] = v + u;
        __syncthreads();
    }
    int run = part[t] - s;                          // exclusive prefix
#pragma unroll
    for (int i = 0; i < 32; ++i) { ofs[t * 32 + i] = run; cur[t * 32 + i] = run; run += loc[i]; }
    if (t == 1023) ofs[NTOT] = run;                 // == NEDGE
}

__global__ void scatter_kernel(const int* __restrict__ col, int* __restrict__ cur,
                               int* __restrict__ rev) {
    const int e = blockIdx.x * 256 + threadIdx.x;
    const int p = atomicAdd(&cur[col[e]], 1);
    rev[p] = e;
}

// ---------------------------------------------------------------------------
// Kernel 3: per-node xW1 = x @ fW1[0:32]  (WMMA, one K-step, no padding waste)
// ---------------------------------------------------------------------------
__global__ __launch_bounds__(256) void xw1_kernel(const float* __restrict__ x,
                                                  const float* __restrict__ fW1,
                                                  float* __restrict__ xw1) {
    __shared__ __align__(32) _Float16 Wt[DH * DIN];   // [n][k], k<32
    for (int i = threadIdx.x; i < DH * DIN; i += 256) {
        const int n = i >> 5, k = i & 31;
        Wt[i] = (_Float16)fW1[k * DH + n];
    }
    __syncthreads();
    const int wave = threadIdx.x >> 5, lane = threadIdx.x & 31;
    const int mrow = lane & 15, kh = lane >> 4;
    const int nodeBase = (blockIdx.x * 8 + wave) * 16;
    const float* xr = x + (size_t)(nodeBase + mrow) * DIN;
    v16h a;
#pragma unroll
    for (int i = 0; i < 8; ++i) a[i]     = (_Float16)xr[kh * 16 + i];
#pragma unroll
    for (int i = 0; i < 8; ++i) a[8 + i] = (_Float16)xr[kh * 16 + 8 + i];
#pragma unroll
    for (int t = 0; t < 4; ++t) {
        v8f c = {};
        const v16h b = *(const v16h*)&Wt[(t * 16 + mrow) * DIN + kh * 16];
        c = __builtin_amdgcn_wmma_f32_16x16x32_f16(false, a, false, b, (short)0, c, false, false);
        const int n = t * 16 + mrow;
#pragma unroll
        for (int r = 0; r < 8; ++r)
            xw1[(size_t)(nodeBase + kh * 8 + r) * DH + n] = c[r];
    }
}

// ---------------------------------------------------------------------------
// Kernel 4: edge layer (h1 rebuilt in-register from xW1 + 3 pos-diff FMAs,
// layer-2 via WMMA) -> coalesced f16 per-edge store to hbuf.
// One wave32 = the 16 edges of one query node.
// ---------------------------------------------------------------------------
__global__ __launch_bounds__(256) void edge_kernel(
    const float* __restrict__ pos, const float* __restrict__ xw1,
    const float* __restrict__ fW1, const float* __restrict__ fb1,
    const float* __restrict__ fW2, const float* __restrict__ fb2,
    const int* __restrict__ col,   _Float16* __restrict__ hbuf) {

    __shared__ __align__(32) _Float16 W2t[DH * DH];   // [n][k]
    __shared__ float w1x[DH], w1y[DH], w1z[DH], b1s[DH], b2s[DH];
    __shared__ __align__(16) _Float16 hst[8 * 16 * DH];

    for (int i = threadIdx.x; i < DH * DH; i += 256) {
        const int n = i >> 6, k = i & 63;
        W2t[i] = (_Float16)fW2[k * DH + n];
    }
    if (threadIdx.x < DH) {
        const int i = threadIdx.x;
        w1x[i] = fW1[(DIN + 0) * DH + i];
        w1y[i] = fW1[(DIN + 1) * DH + i];
        w1z[i] = fW1[(DIN + 2) * DH + i];
        b1s[i] = fb1[i]; b2s[i] = fb2[i];
    }
    __syncthreads();

    const int wave = threadIdx.x >> 5, lane = threadIdx.x & 31;
    const int mrow = lane & 15, kh = lane >> 4;
    const int q = blockIdx.x * 8 + wave, e0 = q * KNN;

    const int nb = col[e0 + mrow];
    const float pdx = pos[q * 3 + 0] - pos[nb * 3 + 0];
    const float pdy = pos[q * 3 + 1] - pos[nb * 3 + 1];
    const float pdz = pos[q * 3 + 2] - pos[nb * 3 + 2];
    const float* xq = xw1 + (size_t)q * DH;

    // layer-1 result directly in A-fragment layout (k == hidden feature idx)
    v16h a2[2];
#pragma unroll
    for (int s = 0; s < 2; ++s) {
#pragma unroll
        for (int i = 0; i < 16; ++i) {
            const int k = s * 32 + kh * 16 + i;
            float v = xq[k] + pdx * w1x[k] + pdy * w1y[k] + pdz * w1z[k] + b1s[k];
            v = v > 0.f ? v : 0.f;
            a2[s][i] = (_Float16)v;
        }
    }

    // layer 2: 4 N-tiles x 2 K-steps of v_wmma_f32_16x16x32_f16
    _Float16* hw = &hst[wave * 16 * DH];
#pragma unroll
    for (int t = 0; t < 4; ++t) {
        v8f c = {};
        const v16h b0 = *(const v16h*)&W2t[(t * 16 + mrow) * DH + kh * 16];
        c = __builtin_amdgcn_wmma_f32_16x16x32_f16(false, a2[0], false, b0, (short)0, c, false, false);
        const v16h b1 = *(const v16h*)&W2t[(t * 16 + mrow) * DH + 32 + kh * 16];
        c = __builtin_amdgcn_wmma_f32_16x16x32_f16(false, a2[1], false, b1, (short)0, c, false, false);
        const int n = t * 16 + mrow;
        const float bias = b2s[n];
#pragma unroll
        for (int r = 0; r < 8; ++r) {
            float v = c[r] + bias; v = v > 0.f ? v : 0.f;
            hw[(kh * 8 + r) * DH + n] = (_Float16)v;   // D row m = r + 8*kh
        }
    }
    __syncthreads();

    // block's 8 waves x 16 edges x 64 halves = 16KB, contiguous in hbuf
    const uint4* src = (const uint4*)hst;
    uint4* dst = (uint4*)(hbuf + (size_t)blockIdx.x * 128 * DH);
    for (int i = threadIdx.x; i < 1024; i += 256) dst[i] = src[i];
}

// ---------------------------------------------------------------------------
// Kernel 5: gather-max over incoming edges. ReLU outputs are >= 0, so u16
// bit order == f16 order -> packed integer max, 16B per edge visit.
// ---------------------------------------------------------------------------
__device__ inline unsigned pkmax_u16(unsigned a, unsigned b) {
    const unsigned lo = ((a & 0xffffu) > (b & 0xffffu)) ? (a & 0xffffu) : (b & 0xffffu);
    const unsigned hi = ((a >> 16)     > (b >> 16))     ? (a & 0xffff0000u) : (b & 0xffff0000u);
    return lo | hi;
}

__global__ __launch_bounds__(256) void agg_kernel(
    const _Float16* __restrict__ hbuf, const int* __restrict__ rev,
    const int* __restrict__ ofs, _Float16* __restrict__ aggh) {
    const int tid  = blockIdx.x * 256 + threadIdx.x;
    const int node = tid >> 3, ch = tid & 7;          // 8 halves per chunk
    const int beg = ofs[node], end = ofs[node + 1];
    const uint4* hb = (const uint4*)hbuf;
    uint4 acc = make_uint4(0u, 0u, 0u, 0u);           // empty segments -> 0.0
    for (int j = beg; j < end; ++j) {
        const int e = rev[j];
        const uint4 v = hb[(size_t)e * 8 + ch];
        acc.x = pkmax_u16(acc.x, v.x); acc.y = pkmax_u16(acc.y, v.y);
        acc.z = pkmax_u16(acc.z, v.z); acc.w = pkmax_u16(acc.w, v.w);
    }
    ((uint4*)aggh)[(size_t)node * 8 + ch] = acc;
}

// ---------------------------------------------------------------------------
// Kernel 6: out = relu(agg @ gW + gb)   [32768,64] @ [64,128], f16 WMMA
// ---------------------------------------------------------------------------
__global__ __launch_bounds__(256) void out_gemm_kernel(
    const _Float16* __restrict__ aggh, const float* __restrict__ gW,
    const float* __restrict__ gb,      float* __restrict__ out) {
    __shared__ __align__(32) _Float16 Wt[DOUT * DH];  // [n][k]
    __shared__ float bs[DOUT];
    for (int i = threadIdx.x; i < DOUT * DH; i += 256) {
        const int n = i >> 6, k = i & 63;
        Wt[i] = (_Float16)gW[k * DOUT + n];
    }
    if (threadIdx.x < DOUT) bs[threadIdx.x] = gb[threadIdx.x];
    __syncthreads();

    const int wave = threadIdx.x >> 5, lane = threadIdx.x & 31;
    const int mrow = lane & 15, kh = lane >> 4;
    const int node0 = (blockIdx.x * 8 + wave) * 16;

    const _Float16* arow = aggh + (size_t)(node0 + mrow) * DH;
    v16h a[2];
#pragma unroll
    for (int s = 0; s < 2; ++s) {
#pragma unroll
        for (int i = 0; i < 8; ++i) a[s][i]     = arow[s * 32 + kh * 16 + i];
#pragma unroll
        for (int i = 0; i < 8; ++i) a[s][8 + i] = arow[s * 32 + kh * 16 + 8 + i];
    }

#pragma unroll
    for (int t = 0; t < 8; ++t) {
        v8f c = {};
        const v16h b0 = *(const v16h*)&Wt[(t * 16 + mrow) * DH + kh * 16];
        c = __builtin_amdgcn_wmma_f32_16x16x32_f16(false, a[0], false, b0, (short)0, c, false, false);
        const v16h b1 = *(const v16h*)&Wt[(t * 16 + mrow) * DH + 32 + kh * 16];
        c = __builtin_amdgcn_wmma_f32_16x16x32_f16(false, a[1], false, b1, (short)0, c, false, false);
        const int n = t * 16 + mrow;
        const float bias = bs[n];
#pragma unroll
        for (int r = 0; r < 8; ++r) {
            float v = c[r] + bias; v = v > 0.f ? v : 0.f;
            out[(size_t)(node0 + kh * 8 + r) * DOUT + n] = v;
        }
    }
}

// ---------------------------------------------------------------------------
extern "C" void kernel_launch(void* const* d_in, const int* in_sizes, int n_in,
                              void* d_out, int out_size, void* d_ws, size_t ws_size,
                              hipStream_t stream) {
    const float* x   = (const float*)d_in[0];
    const float* pos = (const float*)d_in[1];
    // d_in[2] = batch (int32 passthrough)
    const float* fW1 = (const float*)d_in[3];
    const float* fb1 = (const float*)d_in[4];
    const float* fW2 = (const float*)d_in[5];
    const float* fb2 = (const float*)d_in[6];
    const float* gW  = (const float*)d_in[7];
    const float* gb  = (const float*)d_in[8];
    float* out = (float*)d_out;

    char* ws = (char*)d_ws;
    int*      col  = (int*)     (ws + 0x000000);    // NEDGE ints          (2 MB)
    int*      rev  = (int*)     (ws + 0x200000);    // NEDGE ints          (2 MB)
    int*      cnt  = (int*)     (ws + 0x400000);    // NTOT ints           (128 KB)
    int*      ofs  = (int*)     (ws + 0x420000);    // NTOT+1 ints
    int*      cur  = (int*)     (ws + 0x450000);    // NTOT ints
    float*    xw1  = (float*)   (ws + 0x480000);    // NTOT*64 f32         (8 MB)
    _Float16* hbuf = (_Float16*)(ws + 0xC80000);    // NEDGE*64 f16        (64 MB)
    _Float16* aggh = (_Float16*)(ws + 0x4C80000);   // NTOT*64 f16         (4 MB)

    knn_kernel<<<B_GR * (NPER / QPB), 256, 0, stream>>>(pos, col);

    zero_kernel<<<(NTOT / 4 + 255) / 256, 256, 0, stream>>>((uint4*)cnt, NTOT / 4);
    count_kernel<<<NEDGE / 256, 256, 0, stream>>>(col, cnt);
    scan_kernel<<<1, 1024, 0, stream>>>(cnt, ofs, cur);
    scatter_kernel<<<NEDGE / 256, 256, 0, stream>>>(col, cur, rev);

    xw1_kernel<<<NTOT / 128, 256, 0, stream>>>(x, fW1, xw1);
    edge_kernel<<<NEDGE / 128, 256, 0, stream>>>(pos, xw1, fW1, fb1, fW2, fb2, col, hbuf);
    agg_kernel<<<NTOT * 8 / 256, 256, 0, stream>>>(hbuf, rev, ofs, aggh);
    out_gemm_kernel<<<NTOT / 128, 256, 0, stream>>>(aggh, gW, gb, out);

    // tuple passthrough: pos then batch, appended after out
    const size_t outOff = (size_t)NTOT * DOUT;
    hipMemcpyAsync(out + outOff, pos, (size_t)NTOT * 3 * sizeof(float),
                   hipMemcpyDeviceToDevice, stream);
    hipMemcpyAsync(out + outOff + (size_t)NTOT * 3, d_in[2], (size_t)NTOT * sizeof(int),
                   hipMemcpyDeviceToDevice, stream);
}
